// SepNLITBlock_30709016167107
// MI455X (gfx1250) — matrix operand order repaired and checked by reading, not compiled, not run on hardware
//
#include <hip/hip_runtime.h>
#include <hip/hip_bf16.h>

// ---------------- problem constants ----------------
constexpr int Bn = 4, Cn = 64, OUTn = 64, Hn = 256, Wn = 256, M1n = 32, M2n = 32, HIDn = 64;
constexpr int KP = 96;                 // MLP input K padded: 64 ch + 1 coord + 31 zeros
constexpr int PIX = Bn * Hn * Wn;      // 262,144 pixels for W-MLP
constexpr int PH  = Bn * M2n * Hn;     // 32,768 rows for H-MLP

// ---------------- WMMA types / helpers ----------------
typedef __bf16 v16bf __attribute__((ext_vector_type(16)));
typedef __bf16 v8bf  __attribute__((ext_vector_type(8)));
typedef float  v8f   __attribute__((ext_vector_type(8)));

__device__ __forceinline__ v8f wmma_bf16(v16bf a, v16bf b, v8f c) {
    return __builtin_amdgcn_wmma_f32_16x16x32_bf16(false, a, false, b, (short)0, c, false, false);
}

// A-operand (16x32 bf16, M rows x K): per ISA table, lane<16 holds row=lane with
// K chunks {k0..k0+7, k0+16..k0+23}; lane>=16 holds row=lane-16 with {+8, +24}.
__device__ __forceinline__ v16bf loadA16(const __bf16* base, int rs, int k0, int lane) {
    int r  = lane & 15;
    int kb = k0 + ((lane & 16) ? 8 : 0);
    const __bf16* p = base + (size_t)r * rs + kb;
    v8bf lo = *(const v8bf*)p;
    v8bf hi = *(const v8bf*)(p + 16);
    v16bf out;
#pragma unroll
    for (int i = 0; i < 8; ++i) { out[i] = lo[i]; out[i + 8] = hi[i]; }
    return out;
}

// B-operand (32x16 bf16, K x N) from an N-major, K-contiguous buffer (row n = column n):
// lane<16 holds col=lane, K {k0..k0+7, k0+16..k0+23}; lane>=16 holds col=lane-16, K {+8, +24}.
__device__ __forceinline__ v16bf loadB16(const __bf16* Wt, int Kt, int nt, int k0, int lane) {
    int n  = nt * 16 + (lane & 15);
    int kb = k0 + ((lane & 16) ? 8 : 0);
    const __bf16* p = Wt + (size_t)n * Kt + kb;
    v8bf lo = *(const v8bf*)p;
    v8bf hi = *(const v8bf*)(p + 16);
    v16bf out;
#pragma unroll
    for (int i = 0; i < 8; ++i) { out[i] = lo[i]; out[i + 8] = hi[i]; }
    return out;
}

__device__ __forceinline__ v8f bias_bc(const float* b, int nt, int lane) {
    float v = b[nt * 16 + (lane & 15)];
    v8f c;
#pragma unroll
    for (int i = 0; i < 8; ++i) c[i] = v;
    return c;
}

__device__ __forceinline__ float gelu_exact(float v) {
    return 0.5f * v * (1.0f + erff(v * 0.70710678118654752f));
}

// D/C layout: VGPR j, lane<16 -> (row j, col lane); lane>=16 -> (row j+8, col lane-16)
#define D_ROWBASE(lane) (((lane) & 16) ? 8 : 0)

// ---------------- workspace layout (byte offsets) ----------------
constexpr size_t oWW0T = 0;                          // 64x96 bf16
constexpr size_t oWW1T = oWW0T + 12288;              // 64x64 bf16
constexpr size_t oWW2T = oWW1T + 8192;
constexpr size_t oHW0T = oWW2T + 8192;
constexpr size_t oHW1T = oHW0T + 12288;
constexpr size_t oHW2T = oHW1T + 8192;
constexpr size_t oMXB  = oHW2T + 8192;               // mixer  (out,in) bf16
constexpr size_t oSCB  = oMXB  + 8192;               // skip   (out,in) bf16
constexpr size_t oXBF  = oSCB  + 8192;               // (B,C,H,W)  bf16
constexpr size_t oXT   = oXBF  + 33554432ull;        // (B,H,W,96) bf16
constexpr size_t oWBTre= oXT   + 50331648ull;        // (B,H,32,W) bf16
constexpr size_t oWBTim= oWBTre+ 16777216ull;
constexpr size_t oWBPre= oWBTim+ 16777216ull;        // (B,H,W,32) bf16
constexpr size_t oWBPim= oWBPre+ 16777216ull;
constexpr size_t oHIN  = oWBPim+ 16777216ull;        // (B,32,H,96) bf16
constexpr size_t oHBBre= oHIN  + 6291456ull;         // (B,32,32,H) bf16
constexpr size_t oHBBim= oHBBre+ 2097152ull;
constexpr size_t oHBPre= oHBBim+ 2097152ull;         // (B,32,H,32) bf16
constexpr size_t oHBPim= oHBPre+ 2097152ull;
constexpr size_t oTFAre= oHBPim+ 2097152ull;         // tfw (B,32,64,H) bf16
constexpr size_t oTFAim= oTFAre+ 4194304ull;
constexpr size_t oTHre = oTFAim+ 4194304ull;         // tfhw (B,32,64,32) f32
constexpr size_t oTHim = oTHre + 1048576ull;
constexpr size_t oPRre = oTHim + 1048576ull;         // proc (B,32,64,32) bf16
constexpr size_t oPRim = oPRre + 524288ull;
constexpr size_t oRCre = oPRim + 524288ull;          // rec (B,H,64,32) bf16
constexpr size_t oRCim = oRCre + 4194304ull;
constexpr size_t oSP   = oRCim + 4194304ull;         // spatial (B,H,W,64) bf16

// ---------------- kernels ----------------

// K0: convert + transpose all small weight matrices to bf16 N-major/K-contiguous.
__global__ void prep_weights(const float* wW0, const float* wW1, const float* wW2,
                             const float* hW0, const float* hW1, const float* hW2,
                             const float* mxw, const float* scw, char* ws) {
    __bf16* wW0T = (__bf16*)(ws + oWW0T);  __bf16* hW0T = (__bf16*)(ws + oHW0T);
    __bf16* wW1T = (__bf16*)(ws + oWW1T);  __bf16* hW1T = (__bf16*)(ws + oHW1T);
    __bf16* wW2T = (__bf16*)(ws + oWW2T);  __bf16* hW2T = (__bf16*)(ws + oHW2T);
    __bf16* mxb  = (__bf16*)(ws + oMXB);   __bf16* scb  = (__bf16*)(ws + oSCB);
    for (int i = threadIdx.x; i < 64 * 96; i += blockDim.x) {
        int n = i / 96, k = i % 96;
        wW0T[i] = (k < 65) ? (__bf16)wW0[k * 64 + n] : (__bf16)0.0f;
        hW0T[i] = (k < 65) ? (__bf16)hW0[k * 64 + n] : (__bf16)0.0f;
    }
    for (int i = threadIdx.x; i < 64 * 64; i += blockDim.x) {
        int n = i / 64, k = i % 64;
        wW1T[i] = (__bf16)wW1[k * 64 + n];  wW2T[i] = (__bf16)wW2[k * 64 + n];
        hW1T[i] = (__bf16)hW1[k * 64 + n];  hW2T[i] = (__bf16)hW2[k * 64 + n];
        mxb[i]  = (__bf16)mxw[i];           scb[i]  = (__bf16)scw[i];
    }
}

// K1: x(f32 BCHW) -> x_bf (BCHW bf16) and xT (B,H,W,96 bf16 with w-coord @64, zeros above)
__global__ void pack_x(const float* __restrict__ x, char* ws) {
    __bf16* xbf = (__bf16*)(ws + oXBF);
    __bf16* xT  = (__bf16*)(ws + oXT);
    int p = blockIdx.x * blockDim.x + threadIdx.x;       // 0 .. PIX-1
    int b = p >> 16, hw = p & 65535, w = p & 255;
    size_t tbase = (size_t)p * KP;
#pragma unroll 4
    for (int c = 0; c < 64; ++c) {
        size_t src = ((size_t)(b * 64 + c) << 16) + hw;
        float v = x[src];
        xbf[src] = (__bf16)v;
        xT[tbase + c] = (__bf16)v;
    }
    xT[tbase + 64] = (__bf16)(-1.0f + 2.0f * (float)w / 255.0f);
#pragma unroll
    for (int k = 65; k < 96; ++k) xT[tbase + k] = (__bf16)0.0f;
}

// K2: fill h-coordinate / zero padding columns of hin (B,32,H,96)
__global__ void hin_init(char* ws) {
    __bf16* hin = (__bf16*)(ws + oHIN);
    int p = blockIdx.x * blockDim.x + threadIdx.x;       // 0 .. PH-1
    int hh = p & 255;
    size_t base = (size_t)p * KP;
    hin[base + 64] = (__bf16)(-1.0f + 2.0f * (float)hh / 255.0f);
#pragma unroll
    for (int k = 65; k < 96; ++k) hin[base + k] = (__bf16)0.0f;
}

// K3/K5: generic 3-layer complex-output MLP. One wave = 16 rows ("pixels").
// MODE 0: rows are (b,h,w) -> writes wbT (B,H,32,W) and wbP (B,H,W,32)
// MODE 1: rows are (b,m2,h) -> writes hbB (B,32,32,H) and hbP (B,32,H,32)
template <int MODE>
__global__ void __launch_bounds__(128) mlp_kernel(
    const __bf16* __restrict__ X,
    const __bf16* __restrict__ W0T, const float* __restrict__ b0,
    const __bf16* __restrict__ W1T, const float* __restrict__ b1,
    const __bf16* __restrict__ W2T, const float* __restrict__ b2,
    __bf16* __restrict__ o0re, __bf16* __restrict__ o0im,
    __bf16* __restrict__ o1re, __bf16* __restrict__ o1im) {
    __shared__ __bf16 act[4][16 * 64];
    const int lane = threadIdx.x & 31, wid = threadIdx.x >> 5;
    const int group = blockIdx.x * 4 + wid;
    const int rbase = D_ROWBASE(lane);
    __bf16* a = act[wid];
    const __bf16* Xg = X + (size_t)group * 16 * KP;

    // ---- layer 0: (16 x 96) x (96 x 64), GELU ----
#pragma unroll
    for (int t = 0; t < 4; ++t) {
        v8f acc = bias_bc(b0, t, lane);
#pragma unroll
        for (int kc = 0; kc < 3; ++kc)
            acc = wmma_bf16(loadA16(Xg, KP, kc * 32, lane), loadB16(W0T, KP, t, kc * 32, lane), acc);
        int col = t * 16 + (lane & 15);
#pragma unroll
        for (int j = 0; j < 8; ++j) a[(j + rbase) * 64 + col] = (__bf16)gelu_exact(acc[j]);
    }
    // ---- layer 1: (16 x 64) x (64 x 64), GELU ----
    float h1[4][8];
#pragma unroll
    for (int t = 0; t < 4; ++t) {
        v8f acc = bias_bc(b1, t, lane);
#pragma unroll
        for (int kc = 0; kc < 2; ++kc)
            acc = wmma_bf16(loadA16(a, 64, kc * 32, lane), loadB16(W1T, 64, t, kc * 32, lane), acc);
#pragma unroll
        for (int j = 0; j < 8; ++j) h1[t][j] = gelu_exact(acc[j]);
    }
#pragma unroll
    for (int t = 0; t < 4; ++t) {
        int col = t * 16 + (lane & 15);
#pragma unroll
        for (int j = 0; j < 8; ++j) a[(j + rbase) * 64 + col] = (__bf16)h1[t][j];
    }
    // ---- layer 2: (16 x 64) x (64 x 64), no activation; split re/im ----
#pragma unroll
    for (int t = 0; t < 4; ++t) {
        v8f acc = bias_bc(b2, t, lane);
#pragma unroll
        for (int kc = 0; kc < 2; ++kc)
            acc = wmma_bf16(loadA16(a, 64, kc * 32, lane), loadB16(W2T, 64, t, kc * 32, lane), acc);
        int col = t * 16 + (lane & 15);
        int m = col & 31;
        bool isIm = col >= 32;
#pragma unroll
        for (int j = 0; j < 8; ++j) {
            int p = group * 16 + j + rbase;
            __bf16 v = (__bf16)acc[j];
            int i0, i1;
            if (MODE == 0) {
                int b = p >> 16, hh = (p >> 8) & 255, w = p & 255;
                i0 = ((b * 256 + hh) * 32 + m) * 256 + w;   // wbT
                i1 = ((b * 256 + hh) * 256 + w) * 32 + m;   // wbP
            } else {
                int b = p >> 13, mm = (p >> 8) & 31, hh = p & 255;
                i0 = ((b * 32 + mm) * 32 + m) * 256 + hh;   // hbB
                i1 = ((b * 32 + mm) * 256 + hh) * 32 + m;   // hbP
            }
            if (isIm) { o0im[i0] = v; o1im[i1] = v; }
            else      { o0re[i0] = v; o1re[i1] = v; }
        }
    }
}

// K4: tfw[b,c,h,m] = sum_w x[b,c,h,w] * wb[b,h,w,m]   (per (b,h): 64x256 @ 256x32)
__global__ void __launch_bounds__(256) tfw_kernel(char* ws) {
    const __bf16* xbf  = (const __bf16*)(ws + oXBF);
    const __bf16* wbTr = (const __bf16*)(ws + oWBTre);
    const __bf16* wbTi = (const __bf16*)(ws + oWBTim);
    __bf16* tfr = (__bf16*)(ws + oTFAre);
    __bf16* tfi = (__bf16*)(ws + oTFAim);
    __bf16* hin = (__bf16*)(ws + oHIN);
    const int lane = threadIdx.x & 31, wid = threadIdx.x >> 5;
    const int b = blockIdx.x >> 8, h = blockIdx.x & 255;
    const int mt = wid >> 1, nt = wid & 1;
    const int rbase = D_ROWBASE(lane);
    const __bf16* Ab = xbf + ((size_t)(b * 64 + mt * 16) << 16) + (h << 8);   // rs = H*W
    const __bf16* Br = wbTr + (size_t)(b * 256 + h) * 32 * 256;               // Kt = W
    const __bf16* Bi = wbTi + (size_t)(b * 256 + h) * 32 * 256;
    v8f ar = {}, ai = {};
#pragma unroll
    for (int kc = 0; kc < 8; ++kc) {
        v16bf aa = loadA16(Ab, Hn * Wn, kc * 32, lane);
        ar = wmma_bf16(aa, loadB16(Br, 256, nt, kc * 32, lane), ar);
        ai = wmma_bf16(aa, loadB16(Bi, 256, nt, kc * 32, lane), ai);
    }
    int m = nt * 16 + (lane & 15);
#pragma unroll
    for (int j = 0; j < 8; ++j) {
        int c = mt * 16 + j + rbase;
        size_t it = ((size_t)(b * 32 + m) * 64 + c) * 256 + h;      // tfwA (B,32,64,H)
        tfr[it] = (__bf16)ar[j];
        tfi[it] = (__bf16)ai[j];
        hin[((size_t)(b * 32 + m) * 256 + h) * KP + c] = (__bf16)ar[j];
    }
}

// K6: tfhw[b,c,k,m] = sum_h tfw[b,c,h,m] * hb[b,m,h,k]  (per (b,m): 64x256 @ 256x32, cplx)
__global__ void __launch_bounds__(256) tfhw_kernel(char* ws) {
    const __bf16* Are = (const __bf16*)(ws + oTFAre);
    const __bf16* Aim = (const __bf16*)(ws + oTFAim);
    const __bf16* Bre = (const __bf16*)(ws + oHBBre);
    const __bf16* Bim = (const __bf16*)(ws + oHBBim);
    float* Dre = (float*)(ws + oTHre);
    float* Dim = (float*)(ws + oTHim);
    const int lane = threadIdx.x & 31, wid = threadIdx.x >> 5;
    const int b = blockIdx.x >> 5, m = blockIdx.x & 31;
    const int mt = wid >> 1, nt = wid & 1;
    const int rbase = D_ROWBASE(lane);
    const __bf16* Ar = Are + ((size_t)(b * 32 + m) * 64 + mt * 16) * 256;
    const __bf16* Ai = Aim + ((size_t)(b * 32 + m) * 64 + mt * 16) * 256;
    const __bf16* Br = Bre + (size_t)(b * 32 + m) * 32 * 256;
    const __bf16* Bi = Bim + (size_t)(b * 32 + m) * 32 * 256;
    v8f rr = {}, ii = {}, ri = {}, ir = {};
#pragma unroll
    for (int kc = 0; kc < 8; ++kc) {
        v16bf a0 = loadA16(Ar, 256, kc * 32, lane);
        v16bf a1 = loadA16(Ai, 256, kc * 32, lane);
        v16bf b0 = loadB16(Br, 256, nt, kc * 32, lane);
        v16bf b1 = loadB16(Bi, 256, nt, kc * 32, lane);
        rr = wmma_bf16(a0, b0, rr);  ii = wmma_bf16(a1, b1, ii);
        ri = wmma_bf16(a0, b1, ri);  ir = wmma_bf16(a1, b0, ir);
    }
    int kk = nt * 16 + (lane & 15);
#pragma unroll
    for (int j = 0; j < 8; ++j) {
        int c = mt * 16 + j + rbase;
        size_t idx = ((size_t)(b * 32 + m) * 64 + c) * 32 + kk;
        Dre[idx] = rr[j] - ii[j];
        Dim[idx] = ri[j] + ir[j];
    }
}

// K7: proc[b,o,k,m] = sum_i tfhw[b,i,k,m] * Wf[i,o,k,m]   (tiny: VALU)
__global__ void proc_kernel(const float* __restrict__ Wfr, const float* __restrict__ Wfi, char* ws) {
    const float* Tr = (const float*)(ws + oTHre);
    const float* Ti = (const float*)(ws + oTHim);
    __bf16* Pr = (__bf16*)(ws + oPRre);
    __bf16* Pi = (__bf16*)(ws + oPRim);
    int t = blockIdx.x * blockDim.x + threadIdx.x;   // (b,o,kk,m)
    int m = t & 31, kk = (t >> 5) & 31, o = (t >> 10) & 63, b = t >> 16;
    float pr = 0.f, pi = 0.f;
#pragma unroll 4
    for (int i = 0; i < 64; ++i) {
        size_t ti = ((size_t)(b * 32 + m) * 64 + i) * 32 + kk;
        float tr = Tr[ti], tii = Ti[ti];
        size_t wi = (((size_t)i * 64 + o) * 32 + kk) * 32 + m;
        float wr = Wfr[wi], wim = Wfi[wi];
        pr += tr * wr - tii * wim;
        pi += tr * wim + tii * wr;
    }
    size_t idx = ((size_t)(b * 32 + m) * 64 + o) * 32 + kk;   // (B,32,64,32)
    Pr[idx] = (__bf16)pr;
    Pi[idx] = (__bf16)pi;
}

// K8: rec[b,o,h,m] = sum_k proc[b,o,k,m] * conj(hb[b,m,h,k])  (per (b,m): 64x32 @ 32x256)
__global__ void __launch_bounds__(256) rec_kernel(char* ws) {
    const __bf16* Pre = (const __bf16*)(ws + oPRre);
    const __bf16* Pim = (const __bf16*)(ws + oPRim);
    const __bf16* Hre = (const __bf16*)(ws + oHBPre);
    const __bf16* Him = (const __bf16*)(ws + oHBPim);
    __bf16* Rre = (__bf16*)(ws + oRCre);
    __bf16* Rim = (__bf16*)(ws + oRCim);
    const int lane = threadIdx.x & 31, wid = threadIdx.x >> 5;
    const int b = blockIdx.x >> 5, m = blockIdx.x & 31;
    const int rbase = D_ROWBASE(lane);
#pragma unroll
    for (int cc = 0; cc < 8; ++cc) {
        int combo = wid * 8 + cc, mt = combo >> 4, nt = combo & 15;
        v16bf ar = loadA16(Pre + ((size_t)(b * 32 + m) * 64 + mt * 16) * 32, 32, 0, lane);
        v16bf ai = loadA16(Pim + ((size_t)(b * 32 + m) * 64 + mt * 16) * 32, 32, 0, lane);
        v16bf br = loadB16(Hre + (size_t)(b * 32 + m) * 256 * 32, 32, nt, 0, lane);
        v16bf bi = loadB16(Him + (size_t)(b * 32 + m) * 256 * 32, 32, nt, 0, lane);
        v8f z = {};
        v8f rr = wmma_bf16(ar, br, z);
        v8f ii = wmma_bf16(ai, bi, z);
        v8f ir = wmma_bf16(ai, br, z);
        v8f ri = wmma_bf16(ar, bi, z);
        int hh = nt * 16 + (lane & 15);
#pragma unroll
        for (int j = 0; j < 8; ++j) {
            int o = mt * 16 + j + rbase;
            size_t idx = ((size_t)(b * 256 + hh) * 64 + o) * 32 + m;  // (B,H,64,32)
            Rre[idx] = (__bf16)(rr[j] + ii[j]);
            Rim[idx] = (__bf16)(ir[j] - ri[j]);
        }
    }
}

// K9: spatial[b,o,h,w] = Re(sum_m rec * conj(wb)) / (H*W)  (per (b,h): 64x32 @ 32x256)
__global__ void __launch_bounds__(256) spatial_kernel(char* ws) {
    const __bf16* Rre = (const __bf16*)(ws + oRCre);
    const __bf16* Rim = (const __bf16*)(ws + oRCim);
    const __bf16* Wre = (const __bf16*)(ws + oWBPre);
    const __bf16* Wim = (const __bf16*)(ws + oWBPim);
    __bf16* SP = (__bf16*)(ws + oSP);
    const int lane = threadIdx.x & 31, wid = threadIdx.x >> 5;
    const int b = blockIdx.x >> 8, h = blockIdx.x & 255;
    const int rbase = D_ROWBASE(lane);
    const float scale = 1.0f / (float)(Hn * Wn);
#pragma unroll
    for (int cc = 0; cc < 8; ++cc) {
        int combo = wid * 8 + cc, mt = combo >> 4, nt = combo & 15;
        v16bf ar = loadA16(Rre + ((size_t)(b * 256 + h) * 64 + mt * 16) * 32, 32, 0, lane);
        v16bf ai = loadA16(Rim + ((size_t)(b * 256 + h) * 64 + mt * 16) * 32, 32, 0, lane);
        v16bf br = loadB16(Wre + (size_t)(b * 256 + h) * 256 * 32, 32, nt, 0, lane);
        v16bf bi = loadB16(Wim + (size_t)(b * 256 + h) * 256 * 32, 32, nt, 0, lane);
        v8f acc = {};
        acc = wmma_bf16(ar, br, acc);   // re*re
        acc = wmma_bf16(ai, bi, acc);   // + im*im  (conj)
        int w = nt * 16 + (lane & 15);
#pragma unroll
        for (int j = 0; j < 8; ++j) {
            int o = mt * 16 + j + rbase;
            SP[((size_t)(b * 256 + h) * 256 + w) * 64 + o] = (__bf16)(acc[j] * scale);
        }
    }
}

// K10: out = gelu( gelu(spatial @ mixer^T + mb) + x @ sc^T + scb )   one wave = 16 pixels
__global__ void __launch_bounds__(128) final_kernel(const float* __restrict__ mb,
                                                    const float* __restrict__ scbias,
                                                    float* __restrict__ out, char* ws) {
    const __bf16* SP  = (const __bf16*)(ws + oSP);
    const __bf16* XT  = (const __bf16*)(ws + oXT);
    const __bf16* MXB = (const __bf16*)(ws + oMXB);
    const __bf16* SCB = (const __bf16*)(ws + oSCB);
    const int lane = threadIdx.x & 31, wid = threadIdx.x >> 5;
    const int group = blockIdx.x * 4 + wid;
    const int rbase = D_ROWBASE(lane);
    const __bf16* Asp = SP + (size_t)group * 16 * 64;
    const __bf16* Axt = XT + (size_t)group * 16 * KP;
#pragma unroll
    for (int t = 0; t < 4; ++t) {
        v8f a1 = bias_bc(mb, t, lane);
        v8f a2 = bias_bc(scbias, t, lane);
#pragma unroll
        for (int kc = 0; kc < 2; ++kc) {
            a1 = wmma_bf16(loadA16(Asp, 64, kc * 32, lane), loadB16(MXB, 64, t, kc * 32, lane), a1);
            a2 = wmma_bf16(loadA16(Axt, KP, kc * 32, lane), loadB16(SCB, 64, t, kc * 32, lane), a2);
        }
        int o = t * 16 + (lane & 15);
#pragma unroll
        for (int j = 0; j < 8; ++j) {
            int p = group * 16 + j + rbase;
            int b = p >> 16, hh = (p >> 8) & 255, w = p & 255;
            float v = gelu_exact(a1[j]) + a2[j];
            out[(((size_t)b * 64 + o) * 256 + hh) * 256 + w] = gelu_exact(v);
        }
    }
}

// ---------------- launch ----------------
extern "C" void kernel_launch(void* const* d_in, const int* in_sizes, int n_in,
                              void* d_out, int out_size, void* d_ws, size_t ws_size,
                              hipStream_t stream) {
    const float* x    = (const float*)d_in[0];
    const float* wW0  = (const float*)d_in[1];  const float* wb0 = (const float*)d_in[2];
    const float* wW1  = (const float*)d_in[3];  const float* wb1 = (const float*)d_in[4];
    const float* wW2  = (const float*)d_in[5];  const float* wb2 = (const float*)d_in[6];
    const float* hW0  = (const float*)d_in[7];  const float* hb0 = (const float*)d_in[8];
    const float* hW1  = (const float*)d_in[9];  const float* hb1 = (const float*)d_in[10];
    const float* hW2  = (const float*)d_in[11]; const float* hb2 = (const float*)d_in[12];
    const float* Wfr  = (const float*)d_in[13]; const float* Wfi = (const float*)d_in[14];
    const float* mxw  = (const float*)d_in[15]; const float* mxb = (const float*)d_in[16];
    const float* scw  = (const float*)d_in[17]; const float* scb = (const float*)d_in[18];
    float* out = (float*)d_out;
    char* ws = (char*)d_ws;

    prep_weights<<<1, 256, 0, stream>>>(wW0, wW1, wW2, hW0, hW1, hW2, mxw, scw, ws);
    pack_x<<<PIX / 256, 256, 0, stream>>>(x, ws);
    hin_init<<<PH / 256, 256, 0, stream>>>(ws);

    mlp_kernel<0><<<PIX / 64, 128, 0, stream>>>(
        (const __bf16*)(ws + oXT),
        (const __bf16*)(ws + oWW0T), wb0, (const __bf16*)(ws + oWW1T), wb1,
        (const __bf16*)(ws + oWW2T), wb2,
        (__bf16*)(ws + oWBTre), (__bf16*)(ws + oWBTim),
        (__bf16*)(ws + oWBPre), (__bf16*)(ws + oWBPim));

    tfw_kernel<<<Bn * Hn, 256, 0, stream>>>(ws);

    mlp_kernel<1><<<PH / 64, 128, 0, stream>>>(
        (const __bf16*)(ws + oHIN),
        (const __bf16*)(ws + oHW0T), hb0, (const __bf16*)(ws + oHW1T), hb1,
        (const __bf16*)(ws + oHW2T), hb2,
        (__bf16*)(ws + oHBBre), (__bf16*)(ws + oHBBim),
        (__bf16*)(ws + oHBPre), (__bf16*)(ws + oHBPim));

    tfhw_kernel<<<Bn * M2n, 256, 0, stream>>>(ws);
    proc_kernel<<<(Bn * OUTn * M1n * M2n) / 256, 256, 0, stream>>>(Wfr, Wfi, ws);
    rec_kernel<<<Bn * M2n, 256, 0, stream>>>(ws);
    spatial_kernel<<<Bn * Hn, 256, 0, stream>>>(ws);
    final_kernel<<<PIX / 64, 128, 0, stream>>>(mxb, scb, out, ws);
}
